// SplashMHA_69423851372827
// MI455X (gfx1250) — matrix operand order repaired and checked by reading, not compiled, not run on hardware
//
#include <hip/hip_runtime.h>
#include <hip/hip_bf16.h>

typedef _Float16 v16h __attribute__((ext_vector_type(16)));
typedef _Float16 v8h  __attribute__((ext_vector_type(8)));
typedef float    v8f  __attribute__((ext_vector_type(8)));
typedef int      v4i  __attribute__((ext_vector_type(4)));

#ifndef __has_builtin
#define __has_builtin(x) 0
#endif

#if __has_builtin(__builtin_amdgcn_global_load_async_to_lds_b128)
#define HAVE_ASYNC_LDS 1
#else
#define HAVE_ASYNC_LDS 0
#endif

namespace cfg {
constexpr int B  = 2;
constexpr int S  = 4096;
constexpr int D  = 512;
constexpr int H  = 4;
constexpr int HD = 128;
constexpr int NBLK = 8;                              // S / mask block (512)
constexpr float QSCALE = 0.08838834764831845f;       // 128^-0.5
}

// ---------------------------------------------------------------------------
// CDNA5 async global->LDS copy (16B per lane), with portable fallback.
// Builtin prototype (from hipcc diagnostic): (v4i AS1*, v4i AS3*, imm, imm).
// ---------------------------------------------------------------------------
static __device__ __forceinline__ void async_copy16(const _Float16* g, _Float16* l) {
#if HAVE_ASYNC_LDS
    __builtin_amdgcn_global_load_async_to_lds_b128(
        (__attribute__((address_space(1))) v4i*)(uintptr_t)g,
        (__attribute__((address_space(3))) v4i*)(unsigned)(uintptr_t)l, 0, 0);
#else
    *(v8h*)l = *(const v8h*)g;
#endif
}

static __device__ __forceinline__ void wait_async() {
#if HAVE_ASYNC_LDS
#if __has_builtin(__builtin_amdgcn_s_wait_asynccnt)
    __builtin_amdgcn_s_wait_asynccnt(0);
#else
    asm volatile("s_wait_asynccnt 0" ::: "memory");
#endif
#endif
}

// ---------------------------------------------------------------------------
// Fragment helpers (CDNA5 WMMA 16x16x32 f16; A and B both use contiguous k = 16h + s)
// ---------------------------------------------------------------------------
static __device__ __forceinline__ v16h cat8(v8h lo, v8h hi) {
    return __builtin_shufflevector(lo, hi, 0, 1, 2, 3, 4, 5, 6, 7,
                                           8, 9, 10, 11, 12, 13, 14, 15);
}

// A-matrix 16x32 f16: lanes 0-15 -> M=lane, K 0..15;
// lanes 16-31 -> M=lane-16, K 16..31.
static __device__ __forceinline__ v16h load_afrag(const _Float16* __restrict__ base,
                                                  int ld, int m0, int k0, int lane) {
    int m  = m0 + (lane & 15);
    int kh = (lane >> 4) << 4;
    const _Float16* p = base + (size_t)m * ld + k0 + kh;
    return cat8(*(const v8h*)(p), *(const v8h*)(p + 8));
}

// B-matrix 32x16 f16 from K-major storage bt[n][k]: lanes 0-15 -> K 0..15,
// lanes 16-31 -> K 16..31 (16 contiguous halves per lane).
static __device__ __forceinline__ v16h load_bfrag(const _Float16* __restrict__ base,
                                                  int ld, int n0, int k0, int lane) {
    int n = n0 + (lane & 15);
    int k = k0 + ((lane >> 4) << 4);
    return *(const v16h*)(base + (size_t)n * ld + k);
}

static __device__ __forceinline__ v8f wmma(v16h a, v16h b, v8f c) {
    return __builtin_amdgcn_wmma_f32_16x16x32_f16(false, a, false, b,
                                                  (short)0, c, false, false);
}

// ---------------------------------------------------------------------------
// Prep: f32 -> f16 convert, and weight transpose (K-major f16)
// ---------------------------------------------------------------------------
__global__ void cvt_f16_kernel(const float* __restrict__ in,
                               _Float16* __restrict__ out, int n) {
    int i = blockIdx.x * blockDim.x + threadIdx.x;
    if (i < n) out[i] = (_Float16)in[i];
}

__global__ void wtrans_kernel(const float* __restrict__ w,
                              _Float16* __restrict__ wt) {
    int i = blockIdx.x * blockDim.x + threadIdx.x;   // over 512*512
    int k = i >> 9;
    int n = i & 511;
    wt[(size_t)n * 512 + k] = (_Float16)w[i];
}

// ---------------------------------------------------------------------------
// QKV projection: P = X(8192x512) @ Wt; store per-head layouts.
// grid (64, 8, 3), block 256 (8 waves). W tile shared by all 8 waves is
// staged into LDS via double-buffered async copies.
// ---------------------------------------------------------------------------
__global__ __launch_bounds__(256)
void qkv_proj_kernel(const _Float16* __restrict__ x16,
                     const _Float16* __restrict__ kv16,
                     const _Float16* __restrict__ wqt,
                     const _Float16* __restrict__ wkt,
                     const _Float16* __restrict__ wvt,
                     _Float16* __restrict__ q16,
                     _Float16* __restrict__ k16,
                     _Float16* __restrict__ vt16) {
    using namespace cfg;
    __shared__ _Float16 ldsB[2][64][32];               // 8 KB double buffer

    const int mode = blockIdx.z;                       // 0=Q 1=K 2=V
    const _Float16* X  = (mode == 0) ? x16 : kv16;
    const _Float16* Wt = (mode == 0) ? wqt : (mode == 1 ? wkt : wvt);
    const int lane = threadIdx.x & 31;
    const int wav  = threadIdx.x >> 5;
    const int m0   = blockIdx.x * 128 + wav * 16;
    const int nb   = blockIdx.y * 64;

    auto stageB = [&](int buf, int k0) {
        for (int i = threadIdx.x; i < 256; i += 256) { // 64 rows x 4 chunks
            int n = i >> 2, c = i & 3;
            async_copy16(Wt + (size_t)(nb + n) * D + k0 + c * 8,
                         &ldsB[buf][n][c * 8]);
        }
    };

    v8f acc[4] = {};
    stageB(0, 0);
    wait_async();
    __syncthreads();

    for (int kk = 0; kk < 16; ++kk) {
        const int k0 = kk << 5;
        const int cur = kk & 1;
        if (kk + 1 < 16) stageB(cur ^ 1, k0 + 32);

        v16h a = load_afrag(X, D, m0, k0, lane);
#pragma unroll
        for (int j = 0; j < 4; ++j) {
            v16h b = *(const v16h*)&ldsB[cur][j * 16 + (lane & 15)][(lane >> 4) << 4];
            acc[j] = wmma(a, b, acc[j]);
        }
        wait_async();
        __syncthreads();
    }

    const int rowoff = (lane >> 4) << 3;
    const int col    = lane & 15;
#pragma unroll
    for (int j = 0; j < 4; ++j) {
#pragma unroll
        for (int r = 0; r < 8; ++r) {
            int m = m0 + r + rowoff;
            int n = nb + j * 16 + col;
            int b = m >> 12;
            int s = m & (S - 1);
            int h = n >> 7;
            int d = n & (HD - 1);
            float v = acc[j][r];
            if (mode == 0) {
                q16[(((size_t)b * H + h) * S + s) * HD + d] = (_Float16)(v * QSCALE);
            } else if (mode == 1) {
                k16[(((size_t)b * H + h) * S + s) * HD + d] = (_Float16)v;
            } else {
                vt16[(((size_t)b * H + h) * HD + d) * S + s] = (_Float16)v;
            }
        }
    }
}

// ---------------------------------------------------------------------------
// Block-local flash attention. grid (S/128, H, B), block 256 (8 waves).
// K/V chunks (shared by all 8 waves) are staged in LDS with double-buffered
// async copies overlapping the WMMA work.
// ---------------------------------------------------------------------------
__global__ __launch_bounds__(256)
void attn_kernel(const _Float16* __restrict__ q16,
                 const _Float16* __restrict__ k16,
                 const _Float16* __restrict__ vt16,
                 _Float16* __restrict__ o16) {
    using namespace cfg;
    __shared__ _Float16 ldsK[2][32][128];              // 16 KB: keys x d
    __shared__ _Float16 ldsV[2][128][32];              // 16 KB: d x keys
    __shared__ _Float16 plds[8][16][32];               //  8 KB: per-wave P

    const int lane = threadIdx.x & 31;
    const int wav  = threadIdx.x >> 5;
    const int b    = blockIdx.z;
    const int h    = blockIdx.y;
    const int bh   = b * H + h;
    const int q0wg = blockIdx.x * 128;
    const int q0   = q0wg + wav * 16;

    const _Float16* Q  = q16  + (size_t)bh * S * HD;
    const _Float16* Kg = k16  + (size_t)bh * S * HD;
    const _Float16* Vt = vt16 + (size_t)bh * HD * S;

    auto stageKV = [&](int buf, int key0) {
        for (int i = threadIdx.x; i < 512; i += 256) {
            int kr = i >> 4, kc = i & 15;              // 32 rows x 16 chunks
            async_copy16(Kg + (size_t)(key0 + kr) * HD + kc * 8,
                         &ldsK[buf][kr][kc * 8]);
            int vr = i >> 2, vc = i & 3;               // 128 rows x 4 chunks
            async_copy16(Vt + (size_t)vr * S + key0 + vc * 8,
                         &ldsV[buf][vr][vc * 8]);
        }
    };

    v16h qf[4];
#pragma unroll
    for (int dc = 0; dc < 4; ++dc)
        qf[dc] = load_afrag(Q, HD, q0, dc * 32, lane);

    const int qblk = q0wg >> 9;
    const int kbeg = (qblk > 0 ? qblk - 1 : 0) << 9;
    const int kend = ((qblk < NBLK - 1 ? qblk + 1 : NBLK - 1) + 1) << 9;
    const int nIter = (kend - kbeg) >> 5;

    v8f acc[8] = {};
    float mrow[8], lrow[8];
#pragma unroll
    for (int r = 0; r < 8; ++r) { mrow[r] = -1e30f; lrow[r] = 0.f; }

    const int rowoff = (lane >> 4) << 3;
    const int col    = lane & 15;

    stageKV(0, kbeg);
    wait_async();
    __syncthreads();

    for (int it = 0; it < nIter; ++it) {
        const int key0 = kbeg + (it << 5);
        const int cur  = it & 1;
        if (it + 1 < nIter) stageKV(cur ^ 1, key0 + 32);

        // ---- scores: S = Q @ K^T from LDS K tile ------------------------
        v8f sc[2];
        sc[0] = v8f{}; sc[1] = v8f{};
#pragma unroll
        for (int g = 0; g < 2; ++g)
#pragma unroll
            for (int dc = 0; dc < 4; ++dc) {
                v16h kb = *(const v16h*)
                    &ldsK[cur][g * 16 + col][dc * 32 + ((lane >> 4) << 4)];
                sc[g] = wmma(qf[dc], kb, sc[g]);
            }

        // ---- online softmax over this 32-key chunk ----------------------
#pragma unroll
        for (int r = 0; r < 8; ++r) {
            float s0 = sc[0][r], s1 = sc[1][r];
            float mx = fmaxf(s0, s1);
            mx = fmaxf(mx, __shfl_xor(mx, 1, 32));
            mx = fmaxf(mx, __shfl_xor(mx, 2, 32));
            mx = fmaxf(mx, __shfl_xor(mx, 4, 32));
            mx = fmaxf(mx, __shfl_xor(mx, 8, 32));     // row max within half-wave
            float newm  = fmaxf(mrow[r], mx);
            float scale = __expf(mrow[r] - newm);
            float p0 = __expf(s0 - newm);
            float p1 = __expf(s1 - newm);
            float ps = p0 + p1;
            ps += __shfl_xor(ps, 1, 32);
            ps += __shfl_xor(ps, 2, 32);
            ps += __shfl_xor(ps, 4, 32);
            ps += __shfl_xor(ps, 8, 32);
            lrow[r] = lrow[r] * scale + ps;
            mrow[r] = newm;
            sc[0][r] = p0; sc[1][r] = p1;
#pragma unroll
            for (int j = 0; j < 8; ++j) acc[j][r] *= scale;
        }

        // ---- C-layout -> A-layout for P via per-wave LDS tile -----------
#pragma unroll
        for (int r = 0; r < 8; ++r) {
            plds[wav][r + rowoff][col]      = (_Float16)sc[0][r];
            plds[wav][r + rowoff][16 + col] = (_Float16)sc[1][r];
        }
        const _Float16* pp = &plds[wav][lane & 15][(lane >> 4) << 4];
        v16h pf = cat8(*(const v8h*)pp, *(const v8h*)(pp + 8));

        // ---- O += P @ V from LDS V tile ---------------------------------
#pragma unroll
        for (int j = 0; j < 8; ++j) {
            v16h vb = *(const v16h*)
                &ldsV[cur][j * 16 + col][(lane >> 4) << 4];
            acc[j] = wmma(pf, vb, acc[j]);
        }

        wait_async();
        __syncthreads();
    }

    // ---- normalize and store O as [B][S][H*HD] f16 ----------------------
#pragma unroll
    for (int r = 0; r < 8; ++r) {
        float invl = 1.0f / lrow[r];
        int s = q0 + r + rowoff;
#pragma unroll
        for (int j = 0; j < 8; ++j) {
            o16[((size_t)b * S + s) * (H * HD) + h * HD + j * 16 + col] =
                (_Float16)(acc[j][r] * invl);
        }
    }
}

// ---------------------------------------------------------------------------
// Output projection: out = O(8192x512) @ Wo + bo, f32 result.
// grid (64, 8), block 256; Wo tile staged via async LDS double buffer.
// ---------------------------------------------------------------------------
__global__ __launch_bounds__(256)
void oproj_kernel(const _Float16* __restrict__ o16,
                  const _Float16* __restrict__ wot,
                  const float* __restrict__ bo,
                  float* __restrict__ out) {
    using namespace cfg;
    __shared__ _Float16 ldsB[2][64][32];

    const int lane = threadIdx.x & 31;
    const int wav  = threadIdx.x >> 5;
    const int m0   = blockIdx.x * 128 + wav * 16;
    const int nb   = blockIdx.y * 64;

    auto stageB = [&](int buf, int k0) {
        for (int i = threadIdx.x; i < 256; i += 256) {
            int n = i >> 2, c = i & 3;
            async_copy16(wot + (size_t)(nb + n) * D + k0 + c * 8,
                         &ldsB[buf][n][c * 8]);
        }
    };

    v8f acc[4] = {};
    stageB(0, 0);
    wait_async();
    __syncthreads();

    for (int kk = 0; kk < 16; ++kk) {
        const int k0 = kk << 5;
        const int cur = kk & 1;
        if (kk + 1 < 16) stageB(cur ^ 1, k0 + 32);

        v16h a = load_afrag(o16, D, m0, k0, lane);
#pragma unroll
        for (int j = 0; j < 4; ++j) {
            v16h b = *(const v16h*)&ldsB[cur][j * 16 + (lane & 15)][(lane >> 4) << 4];
            acc[j] = wmma(a, b, acc[j]);
        }
        wait_async();
        __syncthreads();
    }

    const int rowoff = (lane >> 4) << 3;
    const int col    = lane & 15;
#pragma unroll
    for (int j = 0; j < 4; ++j) {
#pragma unroll
        for (int r = 0; r < 8; ++r) {
            int m = m0 + r + rowoff;
            int n = nb + j * 16 + col;
            out[(size_t)m * D + n] = acc[j][r] + bo[n];
        }
    }
}

// ---------------------------------------------------------------------------
// Host launcher
// ---------------------------------------------------------------------------
extern "C" void kernel_launch(void* const* d_in, const int* in_sizes, int n_in,
                              void* d_out, int out_size, void* d_ws, size_t ws_size,
                              hipStream_t stream) {
    using namespace cfg;
    const float* q_in  = (const float*)d_in[0];
    const float* kv_in = (const float*)d_in[1];
    const float* Wq    = (const float*)d_in[2];
    const float* Wk    = (const float*)d_in[3];
    const float* Wv    = (const float*)d_in[4];
    const float* Wo    = (const float*)d_in[5];
    const float* bo    = (const float*)d_in[6];
    float* out = (float*)d_out;

    char* ws = (char*)d_ws;
    const size_t szAct = (size_t)B * S * D * sizeof(_Float16);   // 8 MB
    const size_t szW   = (size_t)D * D * sizeof(_Float16);       // 0.5 MB
    _Float16* x16  = (_Float16*)(ws);
    _Float16* kv16 = (_Float16*)(ws + szAct);
    _Float16* wqt  = (_Float16*)(ws + 2 * szAct);
    _Float16* wkt  = (_Float16*)(ws + 2 * szAct + szW);
    _Float16* wvt  = (_Float16*)(ws + 2 * szAct + 2 * szW);
    _Float16* wot  = (_Float16*)(ws + 2 * szAct + 3 * szW);
    _Float16* q16  = (_Float16*)(ws + 2 * szAct + 4 * szW);
    _Float16* k16  = (_Float16*)(ws + 3 * szAct + 4 * szW);
    _Float16* vt16 = (_Float16*)(ws + 4 * szAct + 4 * szW);
    _Float16* o16  = (_Float16*)(ws + 5 * szAct + 4 * szW);

    const int nAct = B * S * D;                                  // 4194304
    cvt_f16_kernel<<<nAct / 256, 256, 0, stream>>>(q_in,  x16,  nAct);
    cvt_f16_kernel<<<nAct / 256, 256, 0, stream>>>(kv_in, kv16, nAct);
    wtrans_kernel<<<(D * D) / 256, 256, 0, stream>>>(Wq, wqt);
    wtrans_kernel<<<(D * D) / 256, 256, 0, stream>>>(Wk, wkt);
    wtrans_kernel<<<(D * D) / 256, 256, 0, stream>>>(Wv, wvt);
    wtrans_kernel<<<(D * D) / 256, 256, 0, stream>>>(Wo, wot);

    qkv_proj_kernel<<<dim3((B * S) / 128, D / 64, 3), 256, 0, stream>>>(
        x16, kv16, wqt, wkt, wvt, q16, k16, vt16);

    attn_kernel<<<dim3(S / 128, H, B), 256, 0, stream>>>(q16, k16, vt16, o16);

    oproj_kernel<<<dim3((B * S) / 128, D / 64), 256, 0, stream>>>(
        o16, wot, bo, out);
}
